// ScaleShiftMACExTB_65111704207446
// MI455X (gfx1250) — matrix operand order repaired and checked by reading, not compiled, not run on hardware
//
#include <hip/hip_runtime.h>
#include <hip/hip_bf16.h>

typedef _Float16 f16;
typedef __attribute__((ext_vector_type(16))) _Float16 v16h;
typedef __attribute__((ext_vector_type(8)))  float    v8f;

#define Nn 10000
#define Ee 160000
#define Cc 64
#define Gg 16
#define NLAY 2
#define PI_F 3.14159265358979f

__device__ __forceinline__ float siluf(float x){ float s=1.f/(1.f+__expf(-x)); return x*s; }
__device__ __forceinline__ float dsiluf(float x){ float s=1.f/(1.f+__expf(-x)); return s*(1.f+x*(1.f-s)); }

__device__ __forceinline__ unsigned ldsoff_u32(const void* p){ return (unsigned)(size_t)p; }

// gfx1250 async global->LDS copy (ASYNCcnt-tracked)
__device__ __forceinline__ void async_ld_b128(unsigned ldsoff, const void* g){
  asm volatile("global_load_async_to_lds_b128 %0, %1, off" :: "v"(ldsoff), "v"(g) : "memory");
}
__device__ __forceinline__ void async_wait0(){
  asm volatile("s_wait_asynccnt 0" ::: "memory");
}

// ---- WMMA fragment helpers (CDNA5 ISA §7.12.2 layouts); A loader works for LDS or global ----
__device__ __forceinline__ v16h frag_A_lds(const f16* t, int ld, int kc, int lane){
  const f16* p = t + (size_t)(lane & 15) * ld + kc * 32 + ((lane & 16) ? 8 : 0);
  v16h a;
#pragma unroll
  for (int j = 0; j < 8; ++j) { a[j] = p[j]; a[j + 8] = p[j + 16]; }
  return a;
}
__device__ __forceinline__ v16h frag_B_g(const f16* Bt, int K, int kc, int nt, int lane){
  const f16* p = Bt + (size_t)(nt * 16 + (lane & 15)) * K + kc * 32 + ((lane & 16) ? 8 : 0);
  v16h b;
#pragma unroll
  for (int j = 0; j < 8; ++j) { b[j] = p[j]; b[j + 8] = p[j + 16]; }
  return b;
}
__device__ __forceinline__ v8f wmma32(v16h a, v16h b, v8f c){
  return __builtin_amdgcn_wmma_f32_16x16x32_f16(false, a, false, b, (short)0, c, false, false);
}

// ---- weight conversion kernels ----
__global__ void k_cvt_t(f16* dst, const float* src, int K, int Ncols, int Kp){
  int idx = blockIdx.x * blockDim.x + threadIdx.x;
  if (idx >= Ncols * Kp) return;
  int n = idx / Kp, k = idx % Kp;
  dst[idx] = (f16)((k < K) ? src[k * Ncols + n] : 0.f);
}
__global__ void k_cvt_c(f16* dst, const float* src, int rows, int cols, int rowsPad){
  int idx = blockIdx.x * blockDim.x + threadIdx.x;
  if (idx >= rowsPad * cols) return;
  int r = idx / cols;
  dst[idx] = (f16)((r < rows) ? src[idx] : 0.f);
}

// ---- geometry ----
__global__ void k_geom(const float* pos, const float* shifts, const int* ei,
                       f16* ef16, float* ys, float* vecb, float* rb){
  int e = blockIdx.x * blockDim.x + threadIdx.x;
  if (e >= Ee) return;
  int s = ei[e], rc = ei[Ee + e];
  float vx = pos[rc*3+0] - pos[s*3+0] + shifts[e*3+0];
  float vy = pos[rc*3+1] - pos[s*3+1] + shifts[e*3+1];
  float vz = pos[rc*3+2] - pos[s*3+2] + shifts[e*3+2];
  float r = sqrtf(vx*vx + vy*vy + vz*vz + 1e-12f);
  vecb[e*3+0]=vx; vecb[e*3+1]=vy; vecb[e*3+2]=vz; rb[e]=r;
  float x = vx/r, y = vy/r, z = vz/r;
  float xr = r / 5.0f, fc = 0.f;
  if (xr < 1.f){
    float xr2=xr*xr, xr4=xr2*xr2, xr5=xr4*xr;
    fc = 1.f - 21.f*xr5 + 35.f*xr5*xr - 15.f*xr5*xr2;
  }
  const float Kb = 0.632455532f;
  for (int b=0;b<8;++b){
    float wb = (b+1)*PI_F/5.0f;
    ef16[(size_t)e*8+b] = (f16)(Kb * sinf(wb*r)/r * fc);
  }
  const float s3=1.73205081f,s5=2.23606798f,s15=3.87298335f,s7=2.64575131f;
  const float s35_8=2.09165007f,s105=10.24695077f,s21_8=1.62018517f;
  float x2=x*x,y2=y*y,z2=z*z;
  float* Y = ys + (size_t)e*16;
  Y[0]=1.f;
  Y[1]=s3*x; Y[2]=s3*y; Y[3]=s3*z;
  Y[4]=s15*x*y; Y[5]=s15*y*z; Y[6]=0.5f*s5*(3.f*z2-1.f); Y[7]=s15*x*z; Y[8]=0.5f*s15*(x2-y2);
  Y[9]=s35_8*y*(3.f*x2-y2); Y[10]=s105*x*y*z; Y[11]=s21_8*y*(5.f*z2-1.f);
  Y[12]=0.5f*s7*z*(5.f*z2-3.f); Y[13]=s21_8*x*(5.f*z2-1.f);
  Y[14]=0.5f*s105*z*(x2-y2); Y[15]=s35_8*x*(x2-3.f*y2);
}

__global__ void k_embed(const float* attrs, const float* Wemb, float* h, f16* h16, int* elemb){
  int idx = blockIdx.x * blockDim.x + threadIdx.x;
  if (idx >= Nn*64) return;
  int n = idx >> 6, c = idx & 63;
  float v = attrs[n*3+0]*Wemb[c] + attrs[n*3+1]*Wemb[64+c] + attrs[n*3+2]*Wemb[128+c];
  h[idx] = v; h16[idx] = (f16)v;
  if (c == 0) elemb[n] = (attrs[n*3+1] > 0.5f) ? 1 : ((attrs[n*3+2] > 0.5f) ? 2 : 0);
}

// ---- hu = h @ Wup (WMMA, async-staged) ----
__global__ void __launch_bounds__(128) k_hu(const f16* h16, const f16* Wupt, float* hu){
  __shared__ f16 buf[4][1024];
  __shared__ float4 obuf4[4][256];
  int lane = threadIdx.x & 31, w = threadIdx.x >> 5;
  int tile = blockIdx.x * 4 + w;
  const int ntiles = Nn / 16;
  bool act = tile < ntiles;
  int t = act ? tile : 0;
  if (act){
    unsigned base = ldsoff_u32(&buf[w][0]);
#pragma unroll
    for (int q = 0; q < 4; ++q){
      int idx = lane + q*32;
      async_ld_b128(base + idx*16, h16 + (size_t)t*1024 + idx*8);
    }
  }
  async_wait0();
  __syncthreads();
  float* obuf = (float*)obuf4[w];
  for (int nt = 0; nt < 4; ++nt){
    v8f acc = {};
    for (int kc = 0; kc < 2; ++kc)
      acc = wmma32(frag_A_lds(buf[w],64,kc,lane), frag_B_g(Wupt,64,kc,nt,lane), acc);
    int cc = nt*16 + (lane & 15);
#pragma unroll
    for (int j = 0; j < 8; ++j){
      int row = ((lane & 16) ? 8 : 0) + j;
      obuf[row*64 + cc] = acc[j];
    }
  }
  __syncthreads();
  if (act){
    for (int idx = lane; idx < 256; idx += 32){
      int row = idx >> 4, seg = idx & 15;
      *(float4*)(hu + (size_t)(t*16+row)*64 + seg*4) = obuf4[w][row*16 + seg];
    }
  }
}

// ---- fused radial MLP (WMMA) + message + scatter ----
__global__ void __launch_bounds__(128) k_edge_fwd(const f16* ef16, const float* ys, const int* ei,
    const float* hu, const f16* W0t, const f16* W1t, const f16* W2t, const f16* W3t, float* abuf){
  __shared__ f16 bA[4][1024];
  __shared__ f16 bB[4][1024];
  int lane = threadIdx.x & 31, w = threadIdx.x >> 5;
  int t = blockIdx.x * 4 + w;
  for (int idx = lane; idx < 512; idx += 32) if ((idx & 31) >= 8) bA[w][idx] = (f16)0.f;
  if (lane < 16) async_ld_b128(ldsoff_u32(&bA[w][0]) + lane*64, ef16 + (size_t)(t*16+lane)*8);
  async_wait0();
  __syncthreads();
  for (int nt = 0; nt < 4; ++nt){
    v8f acc = {};
    acc = wmma32(frag_A_lds(bA[w],32,0,lane), frag_B_g(W0t,32,0,nt,lane), acc);
    int col = nt*16 + (lane & 15);
#pragma unroll
    for (int j = 0; j < 8; ++j){ int row = ((lane&16)?8:0)+j; bB[w][row*64+col] = (f16)siluf(acc[j]); }
  }
  __syncthreads();
  for (int nt = 0; nt < 4; ++nt){
    v8f acc = {};
    for (int kc = 0; kc < 2; ++kc)
      acc = wmma32(frag_A_lds(bB[w],64,kc,lane), frag_B_g(W1t,64,kc,nt,lane), acc);
    int col = nt*16 + (lane & 15);
#pragma unroll
    for (int j = 0; j < 8; ++j){ int row = ((lane&16)?8:0)+j; bA[w][row*64+col] = (f16)siluf(acc[j]); }
  }
  __syncthreads();
  for (int nt = 0; nt < 4; ++nt){
    v8f acc = {};
    for (int kc = 0; kc < 2; ++kc)
      acc = wmma32(frag_A_lds(bA[w],64,kc,lane), frag_B_g(W2t,64,kc,nt,lane), acc);
    int col = nt*16 + (lane & 15);
#pragma unroll
    for (int j = 0; j < 8; ++j){ int row = ((lane&16)?8:0)+j; bB[w][row*64+col] = (f16)siluf(acc[j]); }
  }
  __syncthreads();
  int sr[8], er[8];
#pragma unroll
  for (int j = 0; j < 8; ++j){ int row = ((lane&16)?8:0)+j; int e = t*16+row; sr[j]=ei[e]; er[j]=ei[Ee+e]; }
  for (int nt = 0; nt < 16; ++nt){
    v8f acc = {};
    for (int kc = 0; kc < 2; ++kc)
      acc = wmma32(frag_A_lds(bB[w],64,kc,lane), frag_B_g(W3t,64,kc,nt,lane), acc);
    int l = nt >> 2, moff = l*l, mlen = 2*l+1;
    int c = (nt & 3)*16 + (lane & 15);
#pragma unroll
    for (int j = 0; j < 8; ++j){
      int row = ((lane&16)?8:0)+j; int e = t*16+row;
      float Rh = acc[j] * hu[(size_t)sr[j]*64 + c];
      float* base = abuf + (size_t)er[j]*1024 + c*16 + moff;
      const float* yb = ys + (size_t)e*16 + moff;
      for (int m = 0; m < mlen; ++m) atomicAdd(base + m, Rh * yb[m]);
    }
  }
}

// ---- WMMA linear over (node, m) rows ----
__global__ void __launch_bounds__(128) k_lin_wmma(const float* src, const f16* B4, float* dst, float scale){
  __shared__ f16 At[4][1024];
  int lane = threadIdx.x & 31, w = threadIdx.x >> 5;
  int gw = blockIdx.x * 4 + w;
  int l, tbase;
  if (gw < 625){ l = 0; tbase = 0; }
  else if (gw < 2500){ l = 1; tbase = 625; }
  else if (gw < 5625){ l = 2; tbase = 2500; }
  else { l = 3; tbase = 5625; }
  int tile = gw - tbase;
  int mlen = 2*l + 1, moff = l*l;
  int row0 = tile * 16;
  for (int idx = lane; idx < 1024; idx += 32){
    int row = idx >> 6, c = idx & 63;
    int gr = row0 + row;
    int n = gr / mlen, m = gr - n*mlen;
    At[w][idx] = (f16)src[(size_t)n*1024 + c*16 + moff + m];
  }
  __syncthreads();
  const f16* Bt = B4 + (size_t)l * 4096;
  for (int nt = 0; nt < 4; ++nt){
    v8f acc = {};
    for (int kc = 0; kc < 2; ++kc)
      acc = wmma32(frag_A_lds(At[w],64,kc,lane), frag_B_g(Bt,64,kc,nt,lane), acc);
    int d = nt*16 + (lane & 15);
#pragma unroll
    for (int j = 0; j < 8; ++j){
      int row = ((lane&16)?8:0)+j;
      int gr = row0 + row;
      int n = gr / mlen, m = gr - n*mlen;
      dst[(size_t)n*1024 + d*16 + moff + m] = acc[j] * scale;
    }
  }
}

// ---- feats from A (f16, row-contiguous for WMMA A-frags) ----
__global__ void __launch_bounds__(64) k_feats(const float* A, f16* feats16){
  __shared__ float Ash[1024];
  int n = blockIdx.x, c = threadIdx.x;
  for (int idx = c; idx < 1024; idx += 64) Ash[idx] = A[(size_t)n*1024 + idx];
  __syncthreads();
  float a0 = Ash[c*16];
  f16* fp = feats16 + (size_t)n*576;
  fp[c] = (f16)a0;
  for (int l = 0; l < 4; ++l){
    int moff = l*l, mlen = 2*l+1; float s = 0.f;
    for (int m = 0; m < mlen; ++m){ float v = Ash[c*16+moff+m]; s += v*v; }
    fp[(1+l)*64+c] = (f16)s;
    fp[(5+l)*64+c] = (f16)(a0*s);
  }
}

// ---- h_new = feats@Wprod[el] + h@Wsc[el]  (WMMA over all 3 elements + select) ----
__global__ void __launch_bounds__(128) k_prod_wmma(const f16* feats16, const f16* h16, const int* elemb,
    const f16* Wp0, const f16* Wp1, const f16* Wp2,
    const f16* Ws0, const f16* Ws1, const f16* Ws2,
    float* hout, f16* h16out){
  __shared__ float4 obuf4[4][256];
  int lane = threadIdx.x & 31, w = threadIdx.x >> 5;
  int tile = blockIdx.x * 4 + w;
  const int ntiles = Nn / 16;
  bool act = tile < ntiles;
  int t = act ? tile : 0;
  const f16* fbase = feats16 + (size_t)t*16*576;
  const f16* hbase = h16 + (size_t)t*1024;
  int el[8];
#pragma unroll
  for (int j = 0; j < 8; ++j){ int row = ((lane&16)?8:0)+j; el[j] = elemb[t*16+row]; }
  float* obuf = (float*)obuf4[w];
  for (int nt = 0; nt < 4; ++nt){
    v8f a0 = {}, a1 = {}, a2 = {};
    for (int kc = 0; kc < 18; ++kc){
      v16h a = frag_A_lds(fbase, 576, kc, lane);
      a0 = wmma32(a, frag_B_g(Wp0,576,kc,nt,lane), a0);
      a1 = wmma32(a, frag_B_g(Wp1,576,kc,nt,lane), a1);
      a2 = wmma32(a, frag_B_g(Wp2,576,kc,nt,lane), a2);
    }
    for (int kc = 0; kc < 2; ++kc){
      v16h a = frag_A_lds(hbase, 64, kc, lane);
      a0 = wmma32(a, frag_B_g(Ws0,64,kc,nt,lane), a0);
      a1 = wmma32(a, frag_B_g(Ws1,64,kc,nt,lane), a1);
      a2 = wmma32(a, frag_B_g(Ws2,64,kc,nt,lane), a2);
    }
    int cc = nt*16 + (lane & 15);
#pragma unroll
    for (int j = 0; j < 8; ++j){
      int row = ((lane&16)?8:0)+j;
      float v = (el[j] == 0) ? a0[j] : ((el[j] == 1) ? a1[j] : a2[j]);
      obuf[row*64 + cc] = v;
    }
  }
  __syncthreads();
  if (act){
    for (int idx = lane; idx < 256; idx += 32){
      int row = idx >> 4, seg = idx & 15;
      float4 v = obuf4[w][row*16 + seg];
      size_t o = (size_t)(t*16+row)*64 + seg*4;
      *(float4*)(hout + o) = v;
      h16out[o+0] = (f16)v.x; h16out[o+1] = (f16)v.y;
      h16out[o+2] = (f16)v.z; h16out[o+3] = (f16)v.w;
    }
  }
}

// ---- readout: node_inter accumulation ----
__global__ void __launch_bounds__(64) k_readout(const float* h, const float* w_r0,
    const float* w_r1a, const float* w_r1b, float* ni, int last){
  __shared__ float hn[64], red[64], zsh[16];
  int n = blockIdx.x, c = threadIdx.x;
  hn[c] = h[(size_t)n*64 + c];
  __syncthreads();
  if (last){
    if (c < 16){ float z = 0.f; for (int cc = 0; cc < 64; ++cc) z += hn[cc]*w_r1a[cc*16+c]; zsh[c] = z; }
    __syncthreads();
    if (c == 0){ float e = 0.f; for (int k = 0; k < 16; ++k) e += siluf(zsh[k])*w_r1b[k]; ni[n] += e; }
  } else {
    red[c] = hn[c] * w_r0[c];
    __syncthreads();
    if (c == 0){ float e = 0.f; for (int cc = 0; cc < 64; ++cc) e += red[cc]; ni[n] += e; }
  }
}

__global__ void k_energy(const float* attrs, const float* E0, const float* ni, const int* batch, float* out){
  int n = blockIdx.x * blockDim.x + threadIdx.x;
  if (n >= Nn) return;
  float e = attrs[n*3]*E0[0] + attrs[n*3+1]*E0[1] + attrs[n*3+2]*E0[2] + 0.8f*ni[n] + 0.1f;
  atomicAdd(&out[batch[n]], e);
}

__global__ void __launch_bounds__(64) k_seed(const float* h2, const float* w_r1a, const float* w_r1b, float* g){
  __shared__ float hsh[64], coef[16];
  int n = blockIdx.x, c = threadIdx.x;
  hsh[c] = h2[(size_t)n*64 + c];
  __syncthreads();
  if (c < 16){ float z = 0.f; for (int cc = 0; cc < 64; ++cc) z += hsh[cc]*w_r1a[cc*16+c]; coef[c] = w_r1b[c]*dsiluf(z); }
  __syncthreads();
  float gg = 0.f;
  for (int k = 0; k < 16; ++k) gg += coef[k]*w_r1a[c*16+k];
  g[(size_t)n*64 + c] = 0.8f * gg;
}

// ---- backward node GEMMs: gf = g@Wprod[el]^T (576), gout = g@Wsc[el]^T (64) [WMMA x3 + select] ----
__global__ void __launch_bounds__(128) k_gprod_wmma(const float* gin, const int* elemb,
    const f16* Wpb0, const f16* Wpb1, const f16* Wpb2,
    const f16* Wsb0, const f16* Wsb1, const f16* Wsb2,
    float* gf, float* gout){
  __shared__ f16 buf[4][1024];
  int lane = threadIdx.x & 31, w = threadIdx.x >> 5;
  int tile = blockIdx.x * 4 + w;
  const int ntiles = Nn / 16;
  bool act = tile < ntiles;
  int t = act ? tile : 0;
  for (int idx = lane; idx < 1024; idx += 32){
    int row = idx >> 6, c = idx & 63;
    buf[w][idx] = (f16)(act ? gin[(size_t)(t*16+row)*64+c] : 0.f);
  }
  __syncthreads();
  v16h af0 = frag_A_lds(buf[w],64,0,lane);
  v16h af1 = frag_A_lds(buf[w],64,1,lane);
  int el[8];
#pragma unroll
  for (int j = 0; j < 8; ++j){ int row = ((lane&16)?8:0)+j; el[j] = elemb[t*16+row]; }
  for (int nt = 0; nt < 36; ++nt){
    v8f x0 = {}, x1 = {}, x2 = {};
    x0 = wmma32(af0, frag_B_g(Wpb0,64,0,nt,lane), x0);
    x0 = wmma32(af1, frag_B_g(Wpb0,64,1,nt,lane), x0);
    x1 = wmma32(af0, frag_B_g(Wpb1,64,0,nt,lane), x1);
    x1 = wmma32(af1, frag_B_g(Wpb1,64,1,nt,lane), x1);
    x2 = wmma32(af0, frag_B_g(Wpb2,64,0,nt,lane), x2);
    x2 = wmma32(af1, frag_B_g(Wpb2,64,1,nt,lane), x2);
    int cc = nt*16 + (lane & 15);
#pragma unroll
    for (int j = 0; j < 8; ++j){
      int row = ((lane&16)?8:0)+j;
      float v = (el[j] == 0) ? x0[j] : ((el[j] == 1) ? x1[j] : x2[j]);
      if (act) gf[(size_t)(t*16+row)*576 + cc] = v;
    }
  }
  for (int nt = 0; nt < 4; ++nt){
    v8f x0 = {}, x1 = {}, x2 = {};
    x0 = wmma32(af0, frag_B_g(Wsb0,64,0,nt,lane), x0);
    x0 = wmma32(af1, frag_B_g(Wsb0,64,1,nt,lane), x0);
    x1 = wmma32(af0, frag_B_g(Wsb1,64,0,nt,lane), x1);
    x1 = wmma32(af1, frag_B_g(Wsb1,64,1,nt,lane), x1);
    x2 = wmma32(af0, frag_B_g(Wsb2,64,0,nt,lane), x2);
    x2 = wmma32(af1, frag_B_g(Wsb2,64,1,nt,lane), x2);
    int cc = nt*16 + (lane & 15);
#pragma unroll
    for (int j = 0; j < 8; ++j){
      int row = ((lane&16)?8:0)+j;
      float v = (el[j] == 0) ? x0[j] : ((el[j] == 1) ? x1[j] : x2[j]);
      if (act) gout[(size_t)(t*16+row)*64 + cc] = v;
    }
  }
}

// ---- pointwise gA algebra: gf,A -> gA (to abuf, consumed by k_lin_wmma) ----
__global__ void __launch_bounds__(64) k_node_bwd2(const float* gf, const float* A, float* gAout){
  __shared__ float Ash[1024], gfsh[576];
  int n = blockIdx.x, c = threadIdx.x;
  for (int idx = c; idx < 1024; idx += 64) Ash[idx] = A[(size_t)n*1024 + idx];
  for (int idx = c; idx < 576; idx += 64) gfsh[idx] = gf[(size_t)n*576 + idx];
  __syncthreads();
  float a0 = Ash[c*16];
  float p2[4], gp2[4];
  for (int l = 0; l < 4; ++l){
    int moff = l*l, mlen = 2*l+1; float s = 0.f;
    for (int m = 0; m < mlen; ++m){ float v = Ash[c*16+moff+m]; s += v*v; }
    p2[l] = s;
  }
  float ga0 = gfsh[c];
  for (int l = 0; l < 4; ++l){
    gp2[l] = gfsh[(1+l)*64+c] + gfsh[(5+l)*64+c]*a0;
    ga0 += gfsh[(5+l)*64+c]*p2[l];
  }
  float outv[16];
#pragma unroll
  for (int mm = 0; mm < 16; ++mm){
    int l = (mm < 1) ? 0 : ((mm < 4) ? 1 : ((mm < 9) ? 2 : 3));
    outv[mm] = 2.f*Ash[c*16+mm]*gp2[l];
  }
  outv[0] += ga0;
  float* gp = gAout + (size_t)n*1024 + c*16;
#pragma unroll
  for (int q = 0; q < 4; ++q)
    *(float4*)(gp + 4*q) = make_float4(outv[4*q], outv[4*q+1], outv[4*q+2], outv[4*q+3]);
}

// ---- fused edge backward ----
__global__ void __launch_bounds__(64) k_edge_bwd(const f16* ef16, const float* ys, const float* vecb,
    const float* rb, const int* ei, const float* hu, const float* da,
    const f16* W0t, const f16* W1t, const f16* W2t, const f16* W3t,
    const f16* W0b, const f16* W1b, const f16* W2b, const f16* W3b,
    float* dhu, float* pg){
  __shared__ f16 efb[2][512];
  __shared__ f16 t0s[2][1024], t1s[2][1024], t2s[2][1024];
  __shared__ f16 s0s[2][1024], s1s[2][1024], s2s[2][1024];
  __shared__ f16 Rb[2][4096];
  __shared__ f16 dAs[2][1024], dBs[2][1024];
  __shared__ float dYb[2][256], defb[2][256];
  int lane = threadIdx.x & 31, w = threadIdx.x >> 5;
  int t = blockIdx.x * 2 + w;
  for (int idx = lane; idx < 256; idx += 32) dYb[w][idx] = 0.f;
  for (int idx = lane; idx < 512; idx += 32) if ((idx & 31) >= 8) efb[w][idx] = (f16)0.f;
  if (lane < 16){
    int e = t*16 + lane;
    async_ld_b128(ldsoff_u32(&efb[w][0]) + lane*64, ef16 + (size_t)e*8);
    __builtin_prefetch(da + (size_t)ei[Ee+e]*1024, 0, 3);
  }
  async_wait0();
  __syncthreads();
  for (int nt = 0; nt < 4; ++nt){
    v8f acc = {};
    acc = wmma32(frag_A_lds(efb[w],32,0,lane), frag_B_g(W0t,32,0,nt,lane), acc);
    int col = nt*16 + (lane & 15);
#pragma unroll
    for (int j = 0; j < 8; ++j){ int row=((lane&16)?8:0)+j; float v=acc[j];
      t0s[w][row*64+col]=(f16)siluf(v); s0s[w][row*64+col]=(f16)dsiluf(v); }
  }
  __syncthreads();
  for (int nt = 0; nt < 4; ++nt){
    v8f acc = {};
    for (int kc = 0; kc < 2; ++kc) acc = wmma32(frag_A_lds(t0s[w],64,kc,lane), frag_B_g(W1t,64,kc,nt,lane), acc);
    int col = nt*16 + (lane & 15);
#pragma unroll
    for (int j = 0; j < 8; ++j){ int row=((lane&16)?8:0)+j; float v=acc[j];
      t1s[w][row*64+col]=(f16)siluf(v); s1s[w][row*64+col]=(f16)dsiluf(v); }
  }
  __syncthreads();
  for (int nt = 0; nt < 4; ++nt){
    v8f acc = {};
    for (int kc = 0; kc < 2; ++kc) acc = wmma32(frag_A_lds(t1s[w],64,kc,lane), frag_B_g(W2t,64,kc,nt,lane), acc);
    int col = nt*16 + (lane & 15);
#pragma unroll
    for (int j = 0; j < 8; ++j){ int row=((lane&16)?8:0)+j; float v=acc[j];
      t2s[w][row*64+col]=(f16)siluf(v); s2s[w][row*64+col]=(f16)dsiluf(v); }
  }
  __syncthreads();
  for (int nt = 0; nt < 16; ++nt){
    v8f acc = {};
    for (int kc = 0; kc < 2; ++kc) acc = wmma32(frag_A_lds(t2s[w],64,kc,lane), frag_B_g(W3t,64,kc,nt,lane), acc);
    int col = nt*16 + (lane & 15);
#pragma unroll
    for (int j = 0; j < 8; ++j){ int row=((lane&16)?8:0)+j; Rb[w][row*256+col]=(f16)acc[j]; }
  }
  __syncthreads();
  for (int half = 0; half < 2; ++half){
    int c = lane + half*32;
    for (int row = 0; row < 16; ++row){
      int e = t*16 + row; int snd = ei[e], rcv = ei[Ee+e];
      float hs = hu[(size_t)snd*64 + c];
      float dhs_acc = 0.f;
      for (int l = 0; l < 4; ++l){
        int moff = l*l, mlen = 2*l+1;
        float dav[7]; float S = 0.f;
        for (int m = 0; m < mlen; ++m){
          dav[m] = da[(size_t)rcv*1024 + c*16 + moff + m];
          S += dav[m] * ys[(size_t)e*16 + moff + m];
        }
        float Rv = (float)Rb[w][row*256 + l*64 + c];
        dhs_acc += Rv * S;
        Rb[w][row*256 + l*64 + c] = (f16)(hs * S);
        float Rh = Rv * hs;
        for (int m = 0; m < mlen; ++m) atomicAdd(&dYb[w][row*16 + moff + m], dav[m]*Rh);
      }
      atomicAdd(&dhu[(size_t)snd*64 + c], dhs_acc);
    }
  }
  __syncthreads();
  for (int nt = 0; nt < 4; ++nt){
    v8f acc = {};
    for (int kc = 0; kc < 8; ++kc) acc = wmma32(frag_A_lds(Rb[w],256,kc,lane), frag_B_g(W3b,256,kc,nt,lane), acc);
    int col = nt*16 + (lane & 15);
#pragma unroll
    for (int j = 0; j < 8; ++j){ int row=((lane&16)?8:0)+j;
      dAs[w][row*64+col] = (f16)(acc[j]*(float)s2s[w][row*64+col]); }
  }
  __syncthreads();
  for (int nt = 0; nt < 4; ++nt){
    v8f acc = {};
    for (int kc = 0; kc < 2; ++kc) acc = wmma32(frag_A_lds(dAs[w],64,kc,lane), frag_B_g(W2b,64,kc,nt,lane), acc);
    int col = nt*16 + (lane & 15);
#pragma unroll
    for (int j = 0; j < 8; ++j){ int row=((lane&16)?8:0)+j;
      dBs[w][row*64+col] = (f16)(acc[j]*(float)s1s[w][row*64+col]); }
  }
  __syncthreads();
  for (int nt = 0; nt < 4; ++nt){
    v8f acc = {};
    for (int kc = 0; kc < 2; ++kc) acc = wmma32(frag_A_lds(dBs[w],64,kc,lane), frag_B_g(W1b,64,kc,nt,lane), acc);
    int col = nt*16 + (lane & 15);
#pragma unroll
    for (int j = 0; j < 8; ++j){ int row=((lane&16)?8:0)+j;
      dAs[w][row*64+col] = (f16)(acc[j]*(float)s0s[w][row*64+col]); }
  }
  __syncthreads();
  {
    v8f acc = {};
    for (int kc = 0; kc < 2; ++kc) acc = wmma32(frag_A_lds(dAs[w],64,kc,lane), frag_B_g(W0b,64,kc,0,lane), acc);
    int b = lane & 15;
#pragma unroll
    for (int j = 0; j < 8; ++j){ int row=((lane&16)?8:0)+j; defb[w][row*16+b] = acc[j]; }
  }
  __syncthreads();
  if (lane < 16){
    int row = lane; int e = t*16 + row;
    float vx=vecb[e*3], vy=vecb[e*3+1], vz=vecb[e*3+2], r=rb[e];
    float x=vx/r, y=vy/r, z=vz/r;
    float xr = r/5.0f, fc=0.f, dfc=0.f;
    if (xr < 1.f){
      float xr2=xr*xr, xr4=xr2*xr2, xr5=xr4*xr;
      fc  = 1.f - 21.f*xr5 + 35.f*xr5*xr - 15.f*xr5*xr2;
      dfc = (-105.f*xr4 + 210.f*xr4*xr - 105.f*xr4*xr2) / 5.0f;
    }
    const float Kb = 0.632455532f;
    float gr = 0.f;
    for (int b = 0; b < 8; ++b){
      float wb = (b+1)*PI_F/5.0f;
      float sn = sinf(wb*r), cs = cosf(wb*r);
      float Bv = Kb*sn/r;
      float dB = Kb*(wb*cs/r - sn/(r*r));
      gr += defb[w][row*16+b] * (dB*fc + Bv*dfc);
    }
    const float s3=1.73205081f,s5=2.23606798f,s15=3.87298335f,s7=2.64575131f;
    const float s35_8=2.09165007f,s105=10.24695077f,s21_8=1.62018517f;
    float gy[16];
#pragma unroll
    for (int m = 0; m < 16; ++m) gy[m] = dYb[w][row*16+m];
    float x2=x*x,y2=y*y,z2=z*z;
    float gux = s3*gy[1] + gy[4]*s15*y + gy[7]*s15*z + gy[8]*s15*x
              + gy[9]*s35_8*6.f*x*y + gy[10]*s105*y*z + gy[13]*s21_8*(5.f*z2-1.f)
              + gy[14]*s105*x*z + gy[15]*s35_8*3.f*(x2-y2);
    float guy = s3*gy[2] + gy[4]*s15*x + gy[5]*s15*z - gy[8]*s15*y
              + gy[9]*s35_8*3.f*(x2-y2) + gy[10]*s105*x*z + gy[11]*s21_8*(5.f*z2-1.f)
              - gy[14]*s105*y*z - gy[15]*s35_8*6.f*x*y;
    float guz = s3*gy[3] + gy[5]*s15*y + gy[6]*3.f*s5*z + gy[7]*s15*x
              + gy[10]*s105*x*y + gy[11]*s21_8*10.f*y*z + gy[12]*0.5f*s7*(15.f*z2-3.f)
              + gy[13]*s21_8*10.f*x*z + gy[14]*0.5f*s105*(x2-y2);
    float ud = x*gux + y*guy + z*guz;
    float dvx = gr*x + (gux - x*ud)/r;
    float dvy = gr*y + (guy - y*ud)/r;
    float dvz = gr*z + (guz - z*ud)/r;
    int snd = ei[e], rcv = ei[Ee+e];
    atomicAdd(&pg[rcv*3+0],  dvx); atomicAdd(&pg[rcv*3+1],  dvy); atomicAdd(&pg[rcv*3+2],  dvz);
    atomicAdd(&pg[snd*3+0], -dvx); atomicAdd(&pg[snd*3+1], -dvy); atomicAdd(&pg[snd*3+2], -dvz);
  }
}

// ---- gout += dhu @ Wup^T (+ SCALE*w_r0 bias)  [WMMA] ----
__global__ void __launch_bounds__(128) k_hu_bwd_wmma(const float* dhu, const f16* Wupb,
                                                     const float* w_r0, float* gout, int addr0){
  __shared__ f16 buf[4][1024];
  __shared__ float4 obuf4[4][256];
  int lane = threadIdx.x & 31, w = threadIdx.x >> 5;
  int tile = blockIdx.x * 4 + w;
  const int ntiles = Nn / 16;
  bool act = tile < ntiles;
  int t = act ? tile : 0;
  for (int idx = lane; idx < 1024; idx += 32){
    int row = idx >> 6, c = idx & 63;
    buf[w][idx] = (f16)(act ? dhu[(size_t)(t*16+row)*64+c] : 0.f);
  }
  __syncthreads();
  float* obuf = (float*)obuf4[w];
  for (int nt = 0; nt < 4; ++nt){
    v8f acc = {};
    for (int kc = 0; kc < 2; ++kc)
      acc = wmma32(frag_A_lds(buf[w],64,kc,lane), frag_B_g(Wupb,64,kc,nt,lane), acc);
    int cc = nt*16 + (lane & 15);
#pragma unroll
    for (int j = 0; j < 8; ++j){
      int row = ((lane & 16) ? 8 : 0) + j;
      obuf[row*64 + cc] = acc[j];
    }
  }
  __syncthreads();
  if (act){
    for (int idx = lane; idx < 256; idx += 32){
      int row = idx >> 4, seg = idx & 15;
      float* gp = gout + (size_t)(t*16+row)*64 + seg*4;
      float4 g = *(float4*)gp;
      float4 o = obuf4[w][row*16 + seg];
      float4 b = addr0 ? *(const float4*)(w_r0 + seg*4) : make_float4(0.f,0.f,0.f,0.f);
      g.x += o.x + 0.8f*b.x; g.y += o.y + 0.8f*b.y;
      g.z += o.z + 0.8f*b.z; g.w += o.w + 0.8f*b.w;
      *(float4*)gp = g;
    }
  }
}

__global__ void k_forces(const float* pg, float* out){
  int idx = blockIdx.x * blockDim.x + threadIdx.x;
  if (idx >= Nn*3) return;
  out[Gg + idx] = -pg[idx];
}

extern "C" void kernel_launch(void* const* d_in, const int* in_sizes, int n_in,
                              void* d_out, int out_size, void* d_ws, size_t ws_size,
                              hipStream_t stream){
  (void)in_sizes; (void)n_in; (void)out_size; (void)ws_size;
  const float* pos   = (const float*)d_in[0];
  const float* attrs = (const float*)d_in[1];
  const float* shifts= (const float*)d_in[2];
  const float* Wemb  = (const float*)d_in[3];
  const float* Wup   = (const float*)d_in[4];
  const float* Wlin  = (const float*)d_in[5];
  const float* Wsc   = (const float*)d_in[6];
  const float* r_w0  = (const float*)d_in[7];
  const float* r_w1  = (const float*)d_in[8];
  const float* r_w2  = (const float*)d_in[9];
  const float* r_w3  = (const float*)d_in[10];
  const float* Wprod = (const float*)d_in[11];
  const float* w_r0  = (const float*)d_in[12];
  const float* w_r1a = (const float*)d_in[13];
  const float* w_r1b = (const float*)d_in[14];
  const float* E0    = (const float*)d_in[15];
  const int*   ei    = (const int*)d_in[16];
  const int*   batch = (const int*)d_in[17];
  float* out = (float*)d_out;
  float* ws  = (float*)d_ws;

  size_t off = 0;
  float* ysb  = ws + off; off += (size_t)Ee*16;
  float* vecb = ws + off; off += (size_t)Ee*3;
  float* rb   = ws + off; off += (size_t)Ee;
  float* hb   = ws + off; off += (size_t)3*Nn*64;
  float* hub  = ws + off; off += (size_t)2*Nn*64;
  float* Ab   = ws + off; off += (size_t)2*Nn*1024;
  float* abuf = ws + off; off += (size_t)Nn*1024;
  float* dab  = ws + off; off += (size_t)Nn*1024;
  float* gfb  = ws + off; off += (size_t)Nn*576;
  float* dhub = ws + off; off += (size_t)Nn*64;
  float* gA   = ws + off; off += (size_t)Nn*64;
  float* gB   = ws + off; off += (size_t)Nn*64;
  float* nib  = ws + off; off += (size_t)Nn;
  float* pgb  = ws + off; off += (size_t)Nn*3;
  int*   elemb= (int*)(ws + off); off += (size_t)Nn;
  f16* w16 = (f16*)(ws + off);
  size_t H = 0;
  f16* ef16 = w16 + H; H += (size_t)Ee*8;
  f16* h16b = w16 + H; H += (size_t)3*Nn*64;
  f16* feats16 = w16 + H; H += (size_t)Nn*576;
  f16 *W0t[NLAY], *W0b[NLAY], *W1t[NLAY], *W1b[NLAY], *W2t[NLAY], *W2b[NLAY], *W3t[NLAY], *W3b[NLAY];
  f16 *Wupt[NLAY], *Wupb[NLAY], *Wlt[NLAY], *Wlb[NLAY];
  f16 *Wpt[NLAY][3], *Wpb[NLAY][3], *Wst[NLAY][3], *Wsb[NLAY][3];
  for (int i = 0; i < NLAY; ++i){
    W0t[i]=w16+H; H+=2048;  W0b[i]=w16+H; H+=1024;
    W1t[i]=w16+H; H+=4096;  W1b[i]=w16+H; H+=4096;
    W2t[i]=w16+H; H+=4096;  W2b[i]=w16+H; H+=4096;
    W3t[i]=w16+H; H+=16384; W3b[i]=w16+H; H+=16384;
    Wupt[i]=w16+H; H+=4096; Wupb[i]=w16+H; H+=4096;
    Wlt[i]=w16+H; H+=16384; Wlb[i]=w16+H; H+=16384;
    for (int z = 0; z < 3; ++z){
      Wpt[i][z]=w16+H; H+=36864; Wpb[i][z]=w16+H; H+=36864;
      Wst[i][z]=w16+H; H+=4096;  Wsb[i][z]=w16+H; H+=4096;
    }
  }

  for (int i = 0; i < NLAY; ++i){
    k_cvt_t<<<(64*32+255)/256,256,0,stream>>>(W0t[i], r_w0+(size_t)i*512, 8, 64, 32);
    k_cvt_c<<<(16*64+255)/256,256,0,stream>>>(W0b[i], r_w0+(size_t)i*512, 8, 64, 16);
    k_cvt_t<<<(64*64+255)/256,256,0,stream>>>(W1t[i], r_w1+(size_t)i*4096, 64, 64, 64);
    k_cvt_c<<<(64*64+255)/256,256,0,stream>>>(W1b[i], r_w1+(size_t)i*4096, 64, 64, 64);
    k_cvt_t<<<(64*64+255)/256,256,0,stream>>>(W2t[i], r_w2+(size_t)i*4096, 64, 64, 64);
    k_cvt_c<<<(64*64+255)/256,256,0,stream>>>(W2b[i], r_w2+(size_t)i*4096, 64, 64, 64);
    k_cvt_t<<<(256*64+255)/256,256,0,stream>>>(W3t[i], r_w3+(size_t)i*16384, 64, 256, 64);
    k_cvt_c<<<(64*256+255)/256,256,0,stream>>>(W3b[i], r_w3+(size_t)i*16384, 64, 256, 64);
    k_cvt_t<<<(64*64+255)/256,256,0,stream>>>(Wupt[i], Wup+(size_t)i*4096, 64, 64, 64);
    k_cvt_c<<<(64*64+255)/256,256,0,stream>>>(Wupb[i], Wup+(size_t)i*4096, 64, 64, 64);
    for (int l = 0; l < 4; ++l){
      k_cvt_t<<<(64*64+255)/256,256,0,stream>>>(Wlt[i]+(size_t)l*4096, Wlin+(size_t)(i*4+l)*4096, 64, 64, 64);
      k_cvt_c<<<(64*64+255)/256,256,0,stream>>>(Wlb[i]+(size_t)l*4096, Wlin+(size_t)(i*4+l)*4096, 64, 64, 64);
    }
    for (int z = 0; z < 3; ++z){
      const float* Wp = Wprod + (size_t)(i*3+z)*576*64;
      const float* Ws = Wsc + (size_t)(i*3+z)*4096;
      k_cvt_t<<<(64*576+255)/256,256,0,stream>>>(Wpt[i][z], Wp, 576, 64, 576);
      k_cvt_c<<<(576*64+255)/256,256,0,stream>>>(Wpb[i][z], Wp, 576, 64, 576);
      k_cvt_t<<<(64*64+255)/256,256,0,stream>>>(Wst[i][z], Ws, 64, 64, 64);
      k_cvt_c<<<(64*64+255)/256,256,0,stream>>>(Wsb[i][z], Ws, 64, 64, 64);
    }
  }

  k_geom<<<(Ee+255)/256,256,0,stream>>>(pos, shifts, ei, ef16, ysb, vecb, rb);
  k_embed<<<(Nn*64+255)/256,256,0,stream>>>(attrs, Wemb, hb, h16b, elemb);
  hipMemsetAsync(nib, 0, (size_t)Nn*4, stream);

  for (int i = 0; i < NLAY; ++i){
    k_hu<<<(Nn/16 + 3)/4,128,0,stream>>>(h16b+(size_t)i*Nn*64, Wupt[i], hub+(size_t)i*Nn*64);
    hipMemsetAsync(abuf, 0, (size_t)Nn*1024*4, stream);
    k_edge_fwd<<<Ee/64,128,0,stream>>>(ef16, ysb, ei, hub+(size_t)i*Nn*64,
                                       W0t[i], W1t[i], W2t[i], W3t[i], abuf);
    k_lin_wmma<<<2500,128,0,stream>>>(abuf, Wlt[i], Ab+(size_t)i*Nn*1024, 1.f/16.f);
    k_feats<<<Nn,64,0,stream>>>(Ab+(size_t)i*Nn*1024, feats16);
    k_prod_wmma<<<(Nn/16 + 3)/4,128,0,stream>>>(feats16, h16b+(size_t)i*Nn*64, elemb,
        Wpt[i][0], Wpt[i][1], Wpt[i][2], Wst[i][0], Wst[i][1], Wst[i][2],
        hb+(size_t)(i+1)*Nn*64, h16b+(size_t)(i+1)*Nn*64);
    k_readout<<<Nn,64,0,stream>>>(hb+(size_t)(i+1)*Nn*64, w_r0, w_r1a, w_r1b, nib, (i==NLAY-1)?1:0);
  }
  hipMemsetAsync(out, 0, (size_t)Gg*4, stream);
  k_energy<<<(Nn+255)/256,256,0,stream>>>(attrs, E0, nib, batch, out);

  // ---- backward ----
  hipMemsetAsync(pgb, 0, (size_t)Nn*3*4, stream);
  k_seed<<<Nn,64,0,stream>>>(hb+(size_t)2*Nn*64, w_r1a, w_r1b, gA);
  for (int i = NLAY-1; i >= 0; --i){
    float* gin  = (i == NLAY-1) ? gA : gB;
    float* gout = (i == NLAY-1) ? gB : gA;
    k_gprod_wmma<<<(Nn/16 + 3)/4,128,0,stream>>>(gin, elemb,
        Wpb[i][0], Wpb[i][1], Wpb[i][2], Wsb[i][0], Wsb[i][1], Wsb[i][2], gfb, gout);
    k_node_bwd2<<<Nn,64,0,stream>>>(gfb, Ab+(size_t)i*Nn*1024, abuf);
    k_lin_wmma<<<2500,128,0,stream>>>(abuf, Wlb[i], dab, 1.f/16.f);
    hipMemsetAsync(dhub, 0, (size_t)Nn*64*4, stream);
    k_edge_bwd<<<Ee/32,64,0,stream>>>(ef16, ysb, vecb, rb, ei, hub+(size_t)i*Nn*64, dab,
                                      W0t[i], W1t[i], W2t[i], W3t[i],
                                      W0b[i], W1b[i], W2b[i], W3b[i], dhub, pgb);
    k_hu_bwd_wmma<<<(Nn/16 + 3)/4,128,0,stream>>>(dhub, Wupb[i], w_r0, gout, (i>0)?1:0);
  }
  k_forces<<<(Nn*3+255)/256,256,0,stream>>>(pgb, out);
}
